// TMK_6064493822376
// MI455X (gfx1250) — matrix operand order repaired and checked by reading, not compiled, not run on hardware
//
#include <hip/hip_runtime.h>
#include <hip/hip_bf16.h>

typedef _Float16 v8h  __attribute__((ext_vector_type(8)));
typedef _Float16 v16h __attribute__((ext_vector_type(16)));
typedef float    v8f  __attribute__((ext_vector_type(8)));
typedef int      v4i  __attribute__((ext_vector_type(4)));

#define NROW 4096
#define MDIM 4096
#define DDIM 8

#define BI 128   // block tile output rows (i)
#define BJ 256   // block tile output cols (j)
#define BK 32    // K step (== WMMA K)
#define LDT 40   // padded LDS row stride in f16 (32 data + 8 pad = 80B)

// ---- CDNA5 async global->LDS copy (ASYNCcnt-tracked DMA, no staging VGPRs)
#if defined(__has_builtin)
#if __has_builtin(__builtin_amdgcn_global_load_async_to_lds_b128)
#define TMK_ASYNC_BUILTIN 1
#endif
#if __has_builtin(__builtin_amdgcn_s_wait_asynccnt)
#define TMK_WAIT_BUILTIN 1
#endif
#endif

__device__ __forceinline__ void tmk_async_b128(const _Float16* g, _Float16* l) {
#ifdef TMK_ASYNC_BUILTIN
  typedef __attribute__((address_space(1))) v4i* gp_t;   // global v4i*
  typedef __attribute__((address_space(3))) v4i* lp_t;   // LDS v4i*
  __builtin_amdgcn_global_load_async_to_lds_b128(
      (gp_t)(__UINTPTR_TYPE__)g,
      (lp_t)(unsigned)(__UINTPTR_TYPE__)l, 0, 0);
#else
  unsigned lo = (unsigned)(__UINTPTR_TYPE__)l;   // LDS byte address (low 32 bits)
  asm volatile("global_load_async_to_lds_b128 %0, %1, off"
               :: "v"(lo), "v"(g) : "memory");
#endif
}

__device__ __forceinline__ void tmk_wait_async0() {
#ifdef TMK_WAIT_BUILTIN
  __builtin_amdgcn_s_wait_asynccnt(0);
#else
  asm volatile("s_wait_asynccnt 0" ::: "memory");
#endif
}

// ---------------------------------------------------------------------------
// Phase 1: k_star[i][m] = exp(-sum_d |x[i,d]-grid[m,d]| / ls[d]) stored as f16
// ---------------------------------------------------------------------------
__global__ __launch_bounds__(256) void tmk_kstar_f16(
    const float* __restrict__ x, const float* __restrict__ grid,
    const float* __restrict__ ls, _Float16* __restrict__ A) {
  const int i = blockIdx.x;
  float xr[DDIM], il[DDIM];
#pragma unroll
  for (int d = 0; d < DDIM; ++d) {
    xr[d] = x[i * DDIM + d];
    il[d] = 1.0f / ls[d];
  }
  for (int m = threadIdx.x; m < MDIM; m += 256) {
    float s = 0.f;
#pragma unroll
    for (int d = 0; d < DDIM; ++d)
      s += fabsf(xr[d] - grid[m * DDIM + d]) * il[d];
    A[(size_t)i * MDIM + m] = (_Float16)__expf(-s);
  }
}

// ---------------------------------------------------------------------------
// Phase 2: transpose + convert: Bt[j][k] = (f16) chol_inv[k][j]
// ---------------------------------------------------------------------------
__global__ __launch_bounds__(256) void tmk_transp_f16(
    const float* __restrict__ src,   // [4096 x 4096] (k, j) row-major
    _Float16* __restrict__ dst) {    // [4096 x 4096] (j, k) row-major
  __shared__ float sh[64 * 68];
  const int k0 = blockIdx.y * 64;
  const int j0 = blockIdx.x * 64;
  const int tid = threadIdx.x;
#pragma unroll
  for (int it = 0; it < 4; ++it) {
    int idx = tid + it * 256;
    int r = idx >> 4;
    int c = (idx & 15) * 4;
    float4 v = *(const float4*)(src + (size_t)(k0 + r) * MDIM + j0 + c);
    *(float4*)&sh[r * 68 + c] = v;
  }
  __syncthreads();
#pragma unroll
  for (int it = 0; it < 2; ++it) {
    int idx = tid + it * 256;
    int j  = idx >> 3;
    int kb = (idx & 7) * 8;
    v8h h;
#pragma unroll
    for (int t = 0; t < 8; ++t) h[t] = (_Float16)sh[(kb + t) * 68 + j];
    *(v8h*)(dst + (size_t)(j0 + j) * MDIM + k0 + kb) = h;
  }
}

// ---------------------------------------------------------------------------
// Phase 3: software-pipelined WMMA GEMM with async global->LDS staging.
// 128x256 block tile, double-buffered LDS, one barrier per K-step.
// 8 waves as 2(i) x 4(j); each wave 64x64 = 4x4 fragments, 16 v_wmma / step.
// ---------------------------------------------------------------------------
__global__ __launch_bounds__(256) void tmk_gemm_wmma(
    const _Float16* __restrict__ A,    // [4096 x 4096] (i, k)
    const _Float16* __restrict__ Bt,   // [4096 x 4096] (j, k) pre-transposed
    float* __restrict__ C) {
  __shared__ _Float16 sA[2][BI * LDT];   // 2 x 10240 B
  __shared__ _Float16 sB[2][BJ * LDT];   // 2 x 20480 B

  const int tid  = threadIdx.x;
  const int lane = tid & 31;
  const int wave = tid >> 5;
  const int wm   = wave & 1;
  const int wn   = wave >> 1;
  const int lh   = lane & 15;
  const int sel  = lane >> 4;

  const int i0 = blockIdx.y * BI;
  const int j0 = blockIdx.x * BJ;

  // per-thread staging slots: 16B each; A: 2 chunks, B: 4 chunks
  int rA[2], cA[2], rB[4], cB[4];
#pragma unroll
  for (int it = 0; it < 2; ++it) {
    int e = (tid + it * 256) * 8;
    rA[it] = e >> 5; cA[it] = e & 31;
  }
#pragma unroll
  for (int it = 0; it < 4; ++it) {
    int e = (tid + it * 256) * 8;
    rB[it] = e >> 5; cB[it] = e & 31;
  }

  // ---- prologue: async-stage tile 0 into buffer 0 -------------------------
#pragma unroll
  for (int it = 0; it < 2; ++it)
    tmk_async_b128(A + (size_t)(i0 + rA[it]) * MDIM + cA[it],
                   &sA[0][rA[it] * LDT + cA[it]]);
#pragma unroll
  for (int it = 0; it < 4; ++it)
    tmk_async_b128(Bt + (size_t)(j0 + rB[it]) * MDIM + cB[it],
                   &sB[0][rB[it] * LDT + cB[it]]);
  tmk_wait_async0();
  __syncthreads();

  v8f acc[4][4];
#pragma unroll
  for (int mi = 0; mi < 4; ++mi)
#pragma unroll
    for (int nj = 0; nj < 4; ++nj)
      acc[mi][nj] = {};

  int p = 0;
  for (int kt = 0; kt < MDIM; kt += BK) {
    const bool more = (kt + BK) < MDIM;
    // ---- kick off next tile's DMA into the other buffer -------------------
    if (more) {
      const int kn = kt + BK;
#pragma unroll
      for (int it = 0; it < 2; ++it)
        tmk_async_b128(A + (size_t)(i0 + rA[it]) * MDIM + kn + cA[it],
                       &sA[p ^ 1][rA[it] * LDT + cA[it]]);
#pragma unroll
      for (int it = 0; it < 4; ++it)
        tmk_async_b128(Bt + (size_t)(j0 + rB[it]) * MDIM + kn + cB[it],
                       &sB[p ^ 1][rB[it] * LDT + cB[it]]);
    }
    // ---- fragments from buffer p + 16 WMMAs -------------------------------
    v16h bf[4];
#pragma unroll
    for (int nj = 0; nj < 4; ++nj) {
      const _Float16* q = &sB[p][(wn * 64 + nj * 16 + lh) * LDT + sel * 16];
      v8h lo = *(const v8h*)q;          // K 0..7   / 16..23
      v8h hi = *(const v8h*)(q + 8);    // K 8..15  / 24..31
      bf[nj] = __builtin_shufflevector(lo, hi, 0,1,2,3,4,5,6,7,8,9,10,11,12,13,14,15);
    }
#pragma unroll
    for (int mi = 0; mi < 4; ++mi) {
      const _Float16* q = &sA[p][(wm * 64 + mi * 16 + lh) * LDT + sel * 16];
      v8h lo = *(const v8h*)q;          // K 0..7   / 16..23
      v8h hi = *(const v8h*)(q + 8);    // K 8..15  / 24..31
      v16h af = __builtin_shufflevector(lo, hi, 0,1,2,3,4,5,6,7,8,9,10,11,12,13,14,15);
#pragma unroll
      for (int nj = 0; nj < 4; ++nj)
        acc[mi][nj] = __builtin_amdgcn_wmma_f32_16x16x32_f16(
            false, af, false, bf[nj], (short)0, acc[mi][nj], false, false);
    }
    // ---- own DMAs done, then block-wide visibility ------------------------
    tmk_wait_async0();
    __syncthreads();
    p ^= 1;
  }

  // ---- epilogue: VGPR r -> row r (lanes 0-15) / row 8+r (lanes 16-31) -----
#pragma unroll
  for (int mi = 0; mi < 4; ++mi) {
    const int rowBase = i0 + wm * 64 + mi * 16 + sel * 8;
#pragma unroll
    for (int nj = 0; nj < 4; ++nj) {
      const int col = j0 + wn * 64 + nj * 16 + lh;
#pragma unroll
      for (int r = 0; r < 8; ++r)
        C[(size_t)(rowBase + r) * MDIM + col] = acc[mi][nj][r];
    }
  }
}

// ---------------------------------------------------------------------------
extern "C" void kernel_launch(void* const* d_in, const int* in_sizes, int n_in,
                              void* d_out, int out_size, void* d_ws, size_t ws_size,
                              hipStream_t stream) {
  const float* x    = (const float*)d_in[0];   // [4096, 8]
  const float* grid = (const float*)d_in[1];   // [4096, 8]
  const float* ci   = (const float*)d_in[2];   // [4096, 4096]
  const float* ls   = (const float*)d_in[3];   // [8]
  float* out = (float*)d_out;                  // [4096, 4096] f32

  _Float16* Af = (_Float16*)d_ws;
  _Float16* Bt = (_Float16*)((char*)d_ws + (size_t)NROW * MDIM * sizeof(_Float16));

  tmk_kstar_f16<<<NROW, 256, 0, stream>>>(x, grid, ls, Af);
  tmk_transp_f16<<<dim3(MDIM / 64, MDIM / 64), 256, 0, stream>>>(ci, Bt);

  tmk_gemm_wmma<<<dim3(MDIM / BJ, NROW / BI), 256, 0, stream>>>(Af, Bt, out);
}
